// VQVAE_89850715833079
// MI455X (gfx1250) — matrix operand order repaired and checked by reading, not compiled, not run on hardware
//
#include <hip/hip_runtime.h>

// ---------------------------------------------------------------------------
// VQ-VAE forward for MI455X (gfx1250, wave32, WMMA).
// All conv layers except dec4 via implicit-GEMM with v_wmma_f32_16x16x32_bf16.
// Round 5: stride-2 conv_transpose decomposed into 4 parity-phase convs
// (taps {0,2}/{1} per dim) -> no parity-masked zero K work (4x fewer WMMAs
// in the decoder, which dominates FLOPs). Inter-layer activations bf16.
// ---------------------------------------------------------------------------

typedef __attribute__((ext_vector_type(16))) __bf16 v16bf;
typedef __attribute__((ext_vector_type(2)))  __bf16 bf16x2;
typedef __attribute__((ext_vector_type(8)))  float  v8f;
typedef __attribute__((ext_vector_type(2)))  float  f32x2;

// A-fragment K index for 16-bit A (16x32), contiguous order matching packed B:
// lane half 0: V0..7 -> K 0..15
// lane half 1: V0..7 -> K 16..31
__device__ __forceinline__ constexpr int a_kidx(int v, int half) {
  return 2 * v + 16 * half;
}

// ---------------------------------------------------------------------------
// Weight packing (normal conv): HWIO fp32 -> per-lane bf16 B fragments.
// B (32x16), contiguous k order used here: lane = column n, half selects K 0..15 / 16..31,
// VGPR v holds pair (2v, 2v+1). Layout [kc][nt][v][lane][2] -> coalesced b32.
// ---------------------------------------------------------------------------
template<int KH, int KW, int CIN, int COUT>
__global__ void pack_w(const float* __restrict__ w, __bf16* __restrict__ wpk) {
  constexpr int NT    = COUT / 16;
  constexpr int KREAL = KH * KW * CIN;
  constexpr int KC    = (KREAL + 31) / 32;
  int tid = blockIdx.x * blockDim.x + threadIdx.x;
  if (tid >= KC * NT * 8 * 32) return;
  int lane = tid & 31;
  int v    = (tid >> 5) & 7;
  int nt   = (tid >> 8) % NT;
  int kc   = tid / (256 * NT);
  int half = lane >> 4;
  int co   = nt * 16 + (lane & 15);
  int k0   = 32 * kc + 2 * v + 16 * half;
  __bf16* p = wpk + (size_t)tid * 2;
#pragma unroll
  for (int j = 0; j < 2; ++j) {
    int k = k0 + j;
    float val = 0.f;
    if (k < KREAL) val = w[(size_t)k * COUT + co];   // k=(th*KW+tw)*CIN+ci
    p[j] = (__bf16)val;
  }
}

// Weight packing for one parity phase of a stride-2 3x3 conv_transpose.
// Phase (PH,PW): tap lists th' -> th = PH?1:2*th' (same for w). Unflipped.
template<int CIN, int COUT, int PH, int PW>
__global__ void pack_w_dec(const float* __restrict__ w, __bf16* __restrict__ wpk) {
  constexpr int KHp = PH ? 1 : 2, KWp = PW ? 1 : 2;
  constexpr int NT = COUT / 16;
  constexpr int KC = (KHp * KWp * CIN) / 32;
  int tid = blockIdx.x * blockDim.x + threadIdx.x;
  if (tid >= KC * NT * 256) return;
  int lane = tid & 31;
  int v    = (tid >> 5) & 7;
  int nt   = (tid >> 8) % NT;
  int kc   = tid / (256 * NT);
  int half = lane >> 4;
  int co   = nt * 16 + (lane & 15);
  int k0   = 32 * kc + 2 * v + 16 * half;
  __bf16* p = wpk + (size_t)tid * 2;
#pragma unroll
  for (int j = 0; j < 2; ++j) {
    int k   = k0 + j;
    int tap = k / CIN, ci = k % CIN;
    int thp = tap / KWp, twp = tap % KWp;
    int th  = PH ? 1 : 2 * thp;
    int tw  = PW ? 1 : 2 * twp;
    p[j] = (__bf16)w[((size_t)(th * 3 + tw) * CIN + ci) * COUT + co];
  }
}

// ---------------------------------------------------------------------------
// Implicit-GEMM conv.  One wave: MT M-tiles (MT*16 output pixels along W)
// x 16 output channels; NT waves cover COUT.  Input index:
//   ih = oh*STRIDE + th - PADH,  iw = ow*STRIDE + tw - PADW  (zero outside)
// Output written at physical (OSH*oh+OFH, OSH*ow+OFW) in an
// (OSH*HOUT, OSH*WOUT) surface -> parity-phase stores for deconv (OSH=2).
// ---------------------------------------------------------------------------
template<typename TIN, typename TOUT, int KH, int KW, int CIN, int COUT,
         int STRIDE, int PADH, int PADW, bool LRELU, int MT,
         int OSH, int OFH, int OFW, int HIN, int WIN, int HOUT, int WOUT>
__global__ __launch_bounds__(32 * (COUT / 16))
void conv_wmma(const TIN* __restrict__ x, const __bf16* __restrict__ wpk,
               const float* __restrict__ bias, TOUT* __restrict__ y) {
  constexpr int NT    = COUT / 16;
  constexpr int KREAL = KH * KW * CIN;
  constexpr int KC    = (KREAL + 31) / 32;
  constexpr int WT    = WOUT / (16 * MT);   // M-strips per (logical) row
  constexpr int WP    = OSH * WOUT;         // physical row stride

  const int lane = threadIdx.x & 31;
  const int nt   = threadIdx.x >> 5;
  const int half = lane >> 4;
  const int ln16 = lane & 15;

  int t = blockIdx.x;
  const int owt0 = (t % WT) * MT;  t /= WT;
  const int oh   = t % HOUT;
  const int n    = t / HOUT;

  v8f c[MT];
#pragma unroll
  for (int mt = 0; mt < MT; ++mt) c[mt] = (v8f){0.f,0.f,0.f,0.f,0.f,0.f,0.f,0.f};

  if constexpr (CIN % 32 == 0) {
    constexpr int CC = CIN / 32;       // WMMA chunks per tap
#pragma unroll
    for (int th = 0; th < KH; ++th) {
      const int ih  = oh * STRIDE + th - PADH;             // wave-uniform
      const bool okh = (unsigned)ih < (unsigned)HIN;
#pragma unroll
      for (int tw = 0; tw < KW; ++tw) {
        // per-tap, per-M-tile base pointers & validity (amortized over CC*MT)
        const TIN* base[MT];
        bool ok[MT];
#pragma unroll
        for (int mt = 0; mt < MT; ++mt) {
          const int ow = ((owt0 + mt) << 4) + ln16;
          const int iw = ow * STRIDE + tw - PADW;
          ok[mt]   = okh && (unsigned)iw < (unsigned)WIN;
          base[mt] = x + (size_t)(((n * HIN + ih) * WIN + iw)) * CIN;
        }
#pragma unroll
        for (int cc = 0; cc < CC; ++cc) {
          const int kc = (th * KW + tw) * CC + cc;
          // ---- B fragment: loaded once, reused by MT WMMAs ----
          v16bf b;
          const __bf16* bp = wpk + ((size_t)((kc * NT + nt) * 8) * 32 + lane) * 2;
#pragma unroll
          for (int v = 0; v < 8; ++v) {
            b[2 * v]     = bp[v * 64];
            b[2 * v + 1] = bp[v * 64 + 1];
          }
#pragma unroll
          for (int mt = 0; mt < MT; ++mt) {
            v16bf a;
#pragma unroll
            for (int v = 0; v < 8; ++v) {
              const int ci = 32 * cc + a_kidx(v, half);
              if constexpr (__is_same(TIN, __bf16)) {
                bf16x2 qv = {(__bf16)0.f, (__bf16)0.f};
                if (ok[mt]) qv = *(const bf16x2*)(base[mt] + ci);  // aligned b32
                a[2 * v]     = qv.x;
                a[2 * v + 1] = qv.y;
              } else {
                f32x2 qv = {0.f, 0.f};
                if (ok[mt]) qv = *(const f32x2*)(base[mt] + ci);   // b64 + cvt
                a[2 * v]     = (__bf16)qv.x;
                a[2 * v + 1] = (__bf16)qv.y;
              }
            }
            c[mt] = __builtin_amdgcn_wmma_f32_16x16x32_bf16(
                false, a, false, b, (short)0, c[mt], false, false);
          }
        }
      }
    }
  } else {
    // Generic path (enc1, CIN=3, fp32 input, MT==1): per-element decode.
    static_assert(MT == 1 || CIN % 32 == 0, "generic path expects MT==1");
    const int ow = (owt0 << 4) + ln16;
#pragma unroll
    for (int kc = 0; kc < KC; ++kc) {
      v16bf b;
      const __bf16* bp = wpk + ((size_t)((kc * NT + nt) * 8) * 32 + lane) * 2;
#pragma unroll
      for (int v = 0; v < 8; ++v) {
        b[2 * v]     = bp[v * 64];
        b[2 * v + 1] = bp[v * 64 + 1];
      }
      v16bf a;
#pragma unroll
      for (int v = 0; v < 8; ++v) {
        const int k0 = 32 * kc + a_kidx(v, half);
#pragma unroll
        for (int j = 0; j < 2; ++j) {
          const int k = k0 + j;
          float f = 0.f;
          if (k < KREAL) {
            int tap = k / CIN, ci = k % CIN;
            int th = tap / KW, tw = tap % KW;
            int ih = oh * STRIDE + th - PADH;
            int iw = ow * STRIDE + tw - PADW;
            if ((unsigned)ih < (unsigned)HIN && (unsigned)iw < (unsigned)WIN)
              f = (float)x[(size_t)((n * HIN + ih) * WIN + iw) * CIN + ci];
          }
          a[2 * v + j] = (__bf16)f;
        }
      }
      c[0] = __builtin_amdgcn_wmma_f32_16x16x32_bf16(
          false, a, false, b, (short)0, c[0], false, false);
    }
  }

  // ---- epilogue: C/D layout lane=n (ln16), VGPR r -> m = r + 8*half ----
  const int co = nt * 16 + ln16;
  const float bv = bias[co];
  const int ohp = OSH * oh + OFH;
#pragma unroll
  for (int mt = 0; mt < MT; ++mt) {
#pragma unroll
    for (int r = 0; r < 8; ++r) {
      int owl = ((owt0 + mt) << 4) + r + 8 * half;   // logical column
      int owp = OSH * owl + OFW;                     // physical column
      float vo = c[mt][r] + bv;
      if (LRELU) vo = vo > 0.f ? vo : 0.2f * vo;
      y[(size_t)((n * (OSH * HOUT) + ohp) * WP + owp) * COUT + co] = (TOUT)vo;
    }
  }
}

// ---------------------------------------------------------------------------
// Vector quantize: z [N,32] fp32, codebook [32,64] fp32; argmin L2;
// emit codebook column as bf16 for the decoder.
// ---------------------------------------------------------------------------
__global__ void vq_kernel(const float* __restrict__ z, const float* __restrict__ cb,
                          __bf16* __restrict__ q, int npts) {
  __shared__ float scb[32 * 64];
  for (int i = threadIdx.x; i < 32 * 64; i += blockDim.x) scb[i] = cb[i];
  __syncthreads();
  int p = blockIdx.x * blockDim.x + threadIdx.x;
  if (p >= npts) return;
  float zv[32];
#pragma unroll
  for (int d = 0; d < 32; ++d) zv[d] = z[(size_t)p * 32 + d];
  float best = 3.4e38f; int bi = 0;
  for (int k = 0; k < 64; ++k) {
    float acc = 0.f;
#pragma unroll
    for (int d = 0; d < 32; ++d) { float df = zv[d] - scb[d * 64 + k]; acc += df * df; }
    if (acc < best) { best = acc; bi = k; }   // first-min like jnp.argmin
  }
#pragma unroll
  for (int d = 0; d < 32; ++d) q[(size_t)p * 32 + d] = (__bf16)scb[d * 64 + bi];
}

// ---------------------------------------------------------------------------
// dec4: stride-1 conv_transpose 3x3 SAME == plain pad-1 correlation, Cout=1.
// Memory-bound -> plain fp32 dot kernel over bf16 input.
// ---------------------------------------------------------------------------
__global__ void dec4_kernel(const __bf16* __restrict__ x, const float* __restrict__ w,
                            const float* __restrict__ b, float* __restrict__ y,
                            int H, int W) {
  int idx = blockIdx.x * blockDim.x + threadIdx.x;
  int ow = idx % W; int t = idx / W; int oh = t % H; int n = t / H;
  float acc = b[0];
  for (int th = 0; th < 3; ++th) {
    int ih = oh + th - 1;
    if ((unsigned)ih >= (unsigned)H) continue;
    for (int tw = 0; tw < 3; ++tw) {
      int iw = ow + tw - 1;
      if ((unsigned)iw >= (unsigned)W) continue;
      const __bf16* p = x + (size_t)((n * H + ih) * W + iw) * 32;
      const float* wp = w + (th * 3 + tw) * 32;
#pragma unroll
      for (int ci = 0; ci < 32; ++ci) acc += (float)p[ci] * wp[ci];
    }
  }
  y[idx] = acc;
}

// ---------------------------------------------------------------------------
static constexpr int pack_threads(int kh, int kw, int cin, int cout) {
  return ((kh * kw * cin + 31) / 32) * (cout / 16) * 256;
}

extern "C" void kernel_launch(void* const* d_in, const int* in_sizes, int n_in,
                              void* d_out, int out_size, void* d_ws, size_t ws_size,
                              hipStream_t stream) {
  (void)in_sizes; (void)n_in; (void)out_size; (void)ws_size;

  const float* x    = (const float*)d_in[0];
  const float* e1w  = (const float*)d_in[1];  const float* e1b = (const float*)d_in[2];
  const float* e2w  = (const float*)d_in[3];  const float* e2b = (const float*)d_in[4];
  const float* e3w  = (const float*)d_in[5];  const float* e3b = (const float*)d_in[6];
  const float* e4w  = (const float*)d_in[7];  const float* e4b = (const float*)d_in[8];
  const float* cb   = (const float*)d_in[9];
  const float* d1w  = (const float*)d_in[10]; const float* d1b = (const float*)d_in[11];
  const float* d2w  = (const float*)d_in[12]; const float* d2b = (const float*)d_in[13];
  const float* d3w  = (const float*)d_in[14]; const float* d3b = (const float*)d_in[15];
  const float* d4w  = (const float*)d_in[16]; const float* d4b = (const float*)d_in[17];
  float* out = (float*)d_out;

  // Workspace: ping-pong activation regions + packed weights.
  char*   ws   = (char*)d_ws;
  char*   bufA = ws;                                // h1,h3,q,d2
  char*   bufB = ws + ((size_t)128 << 20);          // h2,z,d1,d3
  __bf16* pk   = (__bf16*)(ws + ((size_t)384 << 20));

  constexpr int PE1 = pack_threads(3,3, 3,32);
  constexpr int PE2 = pack_threads(3,3,32,64);
  constexpr int PE3 = pack_threads(3,3,64,64);
  constexpr int PE4 = pack_threads(1,1,64,32);
  __bf16* pe1 = pk;
  __bf16* pe2 = pe1 + (size_t)PE1 * 2;
  __bf16* pe3 = pe2 + (size_t)PE2 * 2;
  __bf16* pe4 = pe3 + (size_t)PE3 * 2;
  __bf16* pcur = pe4 + (size_t)PE4 * 2;

  pack_w<3,3, 3,32><<<(PE1 + 255) / 256, 256, 0, stream>>>(e1w, pe1);
  pack_w<3,3,32,64><<<(PE2 + 255) / 256, 256, 0, stream>>>(e2w, pe2);
  pack_w<3,3,64,64><<<(PE3 + 255) / 256, 256, 0, stream>>>(e3w, pe3);
  pack_w<1,1,64,32><<<(PE4 + 255) / 256, 256, 0, stream>>>(e4w, pe4);

  // Per-phase deconv packs: 4 phases x 3 layers.
  __bf16* pd[3][4];
  {
    const float* dw[3] = {d1w, d2w, d3w};
    const int cin[3]  = {32, 64, 64};
    const int cout3[3] = {64, 64, 32};
    for (int L = 0; L < 3; ++L)
      for (int ph = 0; ph < 2; ++ph)
        for (int pw2 = 0; pw2 < 2; ++pw2) {
          int thr = pack_threads(ph ? 1 : 2, pw2 ? 1 : 2, cin[L], cout3[L]);
          pd[L][ph * 2 + pw2] = pcur;
          pcur += (size_t)thr * 2;
          (void)dw;
        }
  }
#define LAUNCH_PACK_DEC(L, CI, CO, W_) \
  pack_w_dec<CI,CO,0,0><<<(pack_threads(2,2,CI,CO)+255)/256,256,0,stream>>>(W_, pd[L][0]); \
  pack_w_dec<CI,CO,0,1><<<(pack_threads(2,1,CI,CO)+255)/256,256,0,stream>>>(W_, pd[L][1]); \
  pack_w_dec<CI,CO,1,0><<<(pack_threads(1,2,CI,CO)+255)/256,256,0,stream>>>(W_, pd[L][2]); \
  pack_w_dec<CI,CO,1,1><<<(pack_threads(1,1,CI,CO)+255)/256,256,0,stream>>>(W_, pd[L][3]);
  LAUNCH_PACK_DEC(0, 32, 64, d1w)
  LAUNCH_PACK_DEC(1, 64, 64, d2w)
  LAUNCH_PACK_DEC(2, 64, 32, d3w)
#undef LAUNCH_PACK_DEC

  const int N = 32;
  __bf16* h1 = (__bf16*)bufA;   // 32x128x128x32
  __bf16* h2 = (__bf16*)bufB;   // 32x64x64x64
  __bf16* h3 = (__bf16*)bufA;   // 32x32x32x64
  float*  z  = (float*) bufB;   // 32x32x32x32 fp32 (VQ wants full precision)
  __bf16* q  = (__bf16*)bufA;   // 32x32x32x32
  __bf16* d1 = (__bf16*)bufB;   // 32x64x64x64
  __bf16* d2 = (__bf16*)bufA;   // 32x128x128x64
  __bf16* d3 = (__bf16*)bufB;   // 32x256x256x32

  // encoder (SAME, stride 2 => pad_begin 0 for these even sizes)
  conv_wmma<float ,__bf16,3,3, 3,32, 2,0,0,true ,1, 1,0,0, 256,256,128,128><<<N*128*8,  64, 0, stream>>>(x,  pe1, e1b, h1);
  conv_wmma<__bf16,__bf16,3,3,32,64, 2,0,0,true ,4, 1,0,0, 128,128, 64, 64><<<N* 64*1, 128, 0, stream>>>(h1, pe2, e2b, h2);
  conv_wmma<__bf16,__bf16,3,3,64,64, 2,0,0,true ,2, 1,0,0,  64, 64, 32, 32><<<N* 32*1, 128, 0, stream>>>(h2, pe3, e3b, h3);
  conv_wmma<__bf16,float ,1,1,64,32, 1,0,0,false,2, 1,0,0,  32, 32, 32, 32><<<N* 32*1,  64, 0, stream>>>(h3, pe4, e4b, z);

  // vector quantize: z -> q (bf16)
  vq_kernel<<<(N * 32 * 32) / 128, 128, 0, stream>>>(z, cb, q, N * 32 * 32);

  // decoder: stride-2 transpose as 4 parity-phase convs per layer.
  // Phase (PH,PW): KH'=(PH?1:2), KW'=(PW?1:2), pad=(PH?0:1),(PW?0:1),
  // output at (2*oh+PH, 2*ow+PW).  S = input spatial = logical output.
#define LAUNCH_DEC(L, IN, OUT, CI, CO, S, MT_, BLK) \
  conv_wmma<__bf16,__bf16,2,2,CI,CO,1,1,1,true,MT_, 2,0,0, S,S,S,S><<<N*S*(S/(16*MT_)), BLK, 0, stream>>>(IN, pd[L][0], L==0?d1b:(L==1?d2b:d3b), OUT); \
  conv_wmma<__bf16,__bf16,2,1,CI,CO,1,1,0,true,MT_, 2,0,1, S,S,S,S><<<N*S*(S/(16*MT_)), BLK, 0, stream>>>(IN, pd[L][1], L==0?d1b:(L==1?d2b:d3b), OUT); \
  conv_wmma<__bf16,__bf16,1,2,CI,CO,1,0,1,true,MT_, 2,1,0, S,S,S,S><<<N*S*(S/(16*MT_)), BLK, 0, stream>>>(IN, pd[L][2], L==0?d1b:(L==1?d2b:d3b), OUT); \
  conv_wmma<__bf16,__bf16,1,1,CI,CO,1,0,0,true,MT_, 2,1,1, S,S,S,S><<<N*S*(S/(16*MT_)), BLK, 0, stream>>>(IN, pd[L][3], L==0?d1b:(L==1?d2b:d3b), OUT);
  LAUNCH_DEC(0, q,  d1, 32, 64,  32, 2, 128)   // 32x32 -> 64x64
  LAUNCH_DEC(1, d1, d2, 64, 64,  64, 4, 128)   // 64x64 -> 128x128
  LAUNCH_DEC(2, d2, d3, 64, 32, 128, 4,  64)   // 128x128 -> 256x256
#undef LAUNCH_DEC

  // dec4: stride-1 transpose == pad-1 correlation, Cout=1 -> fp32 dot kernel
  dec4_kernel<<<(N * 256 * 256) / 256, 256, 0, stream>>>(d3, d4w, d4b, out, 256, 256);
}
